// MemorySummaryBank_4767413698779
// MI455X (gfx1250) — matrix operand order repaired and checked by reading, not compiled, not run on hardware
//
#include <hip/hip_runtime.h>
#include <hip/hip_bf16.h>

#define MOMENTUM_F   0.95f
#define ONE_M_M      0.05f
#define TEMP_F       0.35f
#define COS_EPS_F    1e-8f
#define NORM_EPS_F   1e-12f

typedef __attribute__((ext_vector_type(16))) _Float16 v16h;
typedef __attribute__((ext_vector_type(8)))  _Float16 v8h;
typedef __attribute__((ext_vector_type(8)))  float    v8f;

__device__ __forceinline__ float msb_wave_sum(float v) {
#pragma unroll
  for (int off = 16; off; off >>= 1) v += __shfl_xor(v, off, 32);
  return v;
}

// ---------------------------------------------------------------------------
// Kernel 1: sequential slot-routing scan. One workgroup, 32 waves.
// Wave w owns slot w entirely in registers (float4 s[8] <=> k = 4*lane+128*j).
// One __syncthreads per token (double-buffered sims in LDS).
// Dot / norm accumulations use 4 independent chains (dual-issue friendly).
// Epilogue writes sn (f16), slots^T (f16) and the usage flag to workspace.
// ---------------------------------------------------------------------------
__global__ __launch_bounds__(1024)
void msb_update_kernel(const float* __restrict__ wb,
                       const float* __restrict__ slots_in,
                       const float* __restrict__ usage_in, int T,
                       _Float16* __restrict__ sn_out,
                       _Float16* __restrict__ slotsT_out,
                       float* __restrict__ flag_out) {
  __shared__ float2 sbuf[2][32];
  __shared__ float  ubuf[32];

  const int lane = threadIdx.x & 31;
  const int w    = threadIdx.x >> 5;

  float4 s[8];
  const float4* sp = reinterpret_cast<const float4*>(slots_in + (size_t)w * 1024);
#pragma unroll
  for (int j = 0; j < 8; ++j) s[j] = sp[lane + 32 * j];

  float4 sa = make_float4(0.f, 0.f, 0.f, 0.f);
#pragma unroll
  for (int j = 0; j < 8; ++j) {
    sa.x = fmaf(s[j].x, s[j].x, sa.x);
    sa.y = fmaf(s[j].y, s[j].y, sa.y);
    sa.z = fmaf(s[j].z, s[j].z, sa.z);
    sa.w = fmaf(s[j].w, s[j].w, sa.w);
  }
  float snorm = sqrtf(msb_wave_sum((sa.x + sa.y) + (sa.z + sa.w)));
  float u = usage_in[w];

  for (int t = 0; t < T; ++t) {
    const float4* tp = reinterpret_cast<const float4*>(wb + (size_t)t * 1024);
    float4 xv[8];
    float4 da = make_float4(0.f, 0.f, 0.f, 0.f);   // 4 independent dot chains
    float4 na = make_float4(0.f, 0.f, 0.f, 0.f);   // 4 independent norm chains
#pragma unroll
    for (int j = 0; j < 8; ++j) {
      xv[j] = tp[lane + 32 * j];
      da.x = fmaf(s[j].x, xv[j].x, da.x);
      da.y = fmaf(s[j].y, xv[j].y, da.y);
      da.z = fmaf(s[j].z, xv[j].z, da.z);
      da.w = fmaf(s[j].w, xv[j].w, da.w);
      na.x = fmaf(xv[j].x, xv[j].x, na.x);
      na.y = fmaf(xv[j].y, xv[j].y, na.y);
      na.z = fmaf(xv[j].z, xv[j].z, na.z);
      na.w = fmaf(xv[j].w, xv[j].w, na.w);
    }
    float dotp = msb_wave_sum((da.x + da.y) + (da.z + da.w));
    float tn2  = msb_wave_sum((na.x + na.y) + (na.z + na.w));
    float sim = dotp / (fmaxf(sqrtf(tn2), COS_EPS_F) * fmaxf(snorm, COS_EPS_F));
    if (lane == 0) sbuf[t & 1][w] = make_float2(sim, (u == 0.f) ? 1.f : 0.f);
    __syncthreads();

    float2 e  = sbuf[t & 1][lane];
    float key = (e.y != 0.f) ? (1e9f - (float)lane) : e.x;
    float bv = key; int bi = lane;
#pragma unroll
    for (int off = 16; off; off >>= 1) {
      float ov = __shfl_xor(bv, off, 32);
      int   oi = __shfl_xor(bi, off, 32);
      if (ov > bv || (ov == bv && oi < bi)) { bv = ov; bi = oi; }
    }
    if (w == bi) {  // wave-uniform branch: whole owning wave updates its slot
      float4 ra = make_float4(0.f, 0.f, 0.f, 0.f);
#pragma unroll
      for (int j = 0; j < 8; ++j) {
        s[j].x = s[j].x * MOMENTUM_F + xv[j].x * ONE_M_M;
        s[j].y = s[j].y * MOMENTUM_F + xv[j].y * ONE_M_M;
        s[j].z = s[j].z * MOMENTUM_F + xv[j].z * ONE_M_M;
        s[j].w = s[j].w * MOMENTUM_F + xv[j].w * ONE_M_M;
        ra.x = fmaf(s[j].x, s[j].x, ra.x);
        ra.y = fmaf(s[j].y, s[j].y, ra.y);
        ra.z = fmaf(s[j].z, s[j].z, ra.z);
        ra.w = fmaf(s[j].w, s[j].w, ra.w);
      }
      snorm = sqrtf(msb_wave_sum((ra.x + ra.y) + (ra.z + ra.w)));
      u += 1.f;
    }
  }

  // Epilogue: sn (normalized, f16) row-major [slot][k]; slots^T f16 [k][slot].
  float invn = 1.f / fmaxf(snorm, NORM_EPS_F);
#pragma unroll
  for (int j = 0; j < 8; ++j) {
    int k = 4 * lane + 128 * j;
    float v0 = s[j].x, v1 = s[j].y, v2 = s[j].z, v3 = s[j].w;
    sn_out[(size_t)w * 1024 + k + 0] = (_Float16)(v0 * invn);
    sn_out[(size_t)w * 1024 + k + 1] = (_Float16)(v1 * invn);
    sn_out[(size_t)w * 1024 + k + 2] = (_Float16)(v2 * invn);
    sn_out[(size_t)w * 1024 + k + 3] = (_Float16)(v3 * invn);
    slotsT_out[(size_t)(k + 0) * 32 + w] = (_Float16)v0;
    slotsT_out[(size_t)(k + 1) * 32 + w] = (_Float16)v1;
    slotsT_out[(size_t)(k + 2) * 32 + w] = (_Float16)v2;
    slotsT_out[(size_t)(k + 3) * 32 + w] = (_Float16)v3;
  }
  if (lane == 0) ubuf[w] = u;
  __syncthreads();
  if (threadIdx.x == 0) {
    float tsum = 0.f;
    for (int i = 0; i < 32; ++i) tsum += ubuf[i];
    flag_out[0] = (tsum > 0.f) ? 1.f : 0.f;
  }
}

// ---------------------------------------------------------------------------
// Kernel 2: retrieve. 256 threads = 8 waves; each wave owns a 16-token M-tile.
// GEMM1 (scores, K=1024): A = x tokens cast f32->f16 in-register,
//   B = sn^T via contiguous-K loads from sn_f16[slot][k]. 2 WMMA / 32-K chunk.
// Softmax across lanes (N dim of the C fragment) via shfl_xor{1,2,4,8};
//   flag folded into the weights.
// GEMM2 (out, K=32): A = slotsT tile (M = dims), B = weights^T (N = tokens,
//   hoisted single 32B LDS read). C rows are consecutive dims per token ->
//   two b128 stores per lane per 16-dim tile.
// ---------------------------------------------------------------------------
__global__ __launch_bounds__(256)
void msb_retrieve_kernel(const float* __restrict__ x,
                         const _Float16* __restrict__ snh,
                         const _Float16* __restrict__ slotsT,
                         const float* __restrict__ flagp,
                         float* __restrict__ out) {
  __shared__ float    normLds[8][16];
  __shared__ _Float16 wbuf[8][16 * 32];

  const int lane = threadIdx.x & 31;
  const int wave = threadIdx.x >> 5;
  const int m    = lane & 15;     // token row (GEMM1 A) / dim or token column
  const int g    = lane >> 4;     // lane-half selector
  const int tokBase = (blockIdx.x * 8 + wave) * 16;

  const float*    rowp = x + (size_t)(tokBase + m) * 1024 + 16 * g;  // A K-base
  const _Float16* b0p  = snh + (size_t)m * 1024 + 16 * g;            // slots 0-15
  const _Float16* b1p  = snh + (size_t)(16 + m) * 1024 + 16 * g;     // slots 16-31

  v8f acc0 = {}; v8f acc1 = {};
  float4 nv = make_float4(0.f, 0.f, 0.f, 0.f);   // 4 independent norm chains

#pragma unroll 2
  for (int k0 = 0; k0 < 1024; k0 += 32) {
    __builtin_prefetch(rowp + k0 + 64, 0, 1);   // global_prefetch_b8, 2 chunks ahead
    const float4* p = reinterpret_cast<const float4*>(rowp + k0);
    float4 a0 = p[0];      // K = kb+0..3
    float4 a1 = p[1];      // K = kb+4..7
    float4 c0 = p[2];      // K = kb+8..11
    float4 c1 = p[3];      // K = kb+12..15
    nv.x = fmaf(a0.x, a0.x, fmaf(a1.x, a1.x, fmaf(c0.x, c0.x, fmaf(c1.x, c1.x, nv.x))));
    nv.y = fmaf(a0.y, a0.y, fmaf(a1.y, a1.y, fmaf(c0.y, c0.y, fmaf(c1.y, c1.y, nv.y))));
    nv.z = fmaf(a0.z, a0.z, fmaf(a1.z, a1.z, fmaf(c0.z, c0.z, fmaf(c1.z, c1.z, nv.z))));
    nv.w = fmaf(a0.w, a0.w, fmaf(a1.w, a1.w, fmaf(c0.w, c0.w, fmaf(c1.w, c1.w, nv.w))));
    v16h A;
    A[0]  = (_Float16)a0.x; A[1]  = (_Float16)a0.y; A[2]  = (_Float16)a0.z; A[3]  = (_Float16)a0.w;
    A[4]  = (_Float16)a1.x; A[5]  = (_Float16)a1.y; A[6]  = (_Float16)a1.z; A[7]  = (_Float16)a1.w;
    A[8]  = (_Float16)c0.x; A[9]  = (_Float16)c0.y; A[10] = (_Float16)c0.z; A[11] = (_Float16)c0.w;
    A[12] = (_Float16)c1.x; A[13] = (_Float16)c1.y; A[14] = (_Float16)c1.z; A[15] = (_Float16)c1.w;

    v16h B0 = *reinterpret_cast<const v16h*>(b0p + k0);
    v16h B1 = *reinterpret_cast<const v16h*>(b1p + k0);
    acc0 = __builtin_amdgcn_wmma_f32_16x16x32_f16(false, A, false, B0, (short)0, acc0, false, false);
    acc1 = __builtin_amdgcn_wmma_f32_16x16x32_f16(false, A, false, B1, (short)0, acc1, false, false);
  }
  float n2 = (nv.x + nv.y) + (nv.z + nv.w);

  // Per-token ||x||^2: lanes m and m+16 partition K; combine across halves.
  n2 += __shfl_xor(n2, 16, 32);
  if (g == 0) normLds[wave][m] = n2;
  __syncthreads();
  const float4* nlp = reinterpret_cast<const float4*>(&normLds[wave][8 * g]);
  float4 nA = nlp[0], nB = nlp[1];
  float nr[8] = {nA.x, nA.y, nA.z, nA.w, nB.x, nB.y, nB.z, nB.w};

  float flag = *flagp;  // 0 when bank empty -> zero weights -> zero output

  // Softmax over the 32 slots (lanes hold N; rows r hold tokens r + 8g).
#pragma unroll
  for (int r = 0; r < 8; ++r) {
    float fac = 1.f / (fmaxf(sqrtf(nr[r]), NORM_EPS_F) * TEMP_F);
    float s0v = acc0[r] * fac;
    float s1v = acc1[r] * fac;
    float mx = fmaxf(s0v, s1v);
    mx = fmaxf(mx, __shfl_xor(mx, 1, 32));
    mx = fmaxf(mx, __shfl_xor(mx, 2, 32));
    mx = fmaxf(mx, __shfl_xor(mx, 4, 32));
    mx = fmaxf(mx, __shfl_xor(mx, 8, 32));
    float e0 = __expf(s0v - mx);
    float e1 = __expf(s1v - mx);
    float sm = e0 + e1;
    sm += __shfl_xor(sm, 1, 32);
    sm += __shfl_xor(sm, 2, 32);
    sm += __shfl_xor(sm, 4, 32);
    sm += __shfl_xor(sm, 8, 32);
    float rinv = flag / sm;
    int t = r + 8 * g;
    wbuf[wave][t * 32 + m]      = (_Float16)(e0 * rinv);
    wbuf[wave][t * 32 + 16 + m] = (_Float16)(e1 * rinv);
  }
  __syncthreads();

  // GEMM2: B = weights^T -> wbuf[token][slot], contiguous slot-K per lane.
  // Loop-invariant: one 32B LDS read serves all 64 dim (M) tiles.
  v16h B2 = *reinterpret_cast<const v16h*>(&wbuf[wave][m * 32 + 16 * g]);
  float* orow = out + (size_t)(tokBase + m) * 1024 + 8 * g;

#pragma unroll 4
  for (int d0 = 0; d0 < 1024; d0 += 16) {
    // A = slotsT tile: lane holds dim d0+m, slot-K runs 16g..16g+15.
    const _Float16* ap = slotsT + (size_t)(d0 + m) * 32 + 16 * g;
    union { v16h v; v8h h[2]; } ua;
    ua.h[0] = *reinterpret_cast<const v8h*>(ap);
    ua.h[1] = *reinterpret_cast<const v8h*>(ap + 8);
    v8f c = {};
    c = __builtin_amdgcn_wmma_f32_16x16x32_f16(false, ua.v, false, B2, (short)0, c, false, false);
    // C: lane = token m, VGPR r = dim d0 + 8g + r -> 8 consecutive floats.
    float4 lo = make_float4(c[0], c[1], c[2], c[3]);
    float4 hi = make_float4(c[4], c[5], c[6], c[7]);
    reinterpret_cast<float4*>(orow + d0)[0] = lo;
    reinterpret_cast<float4*>(orow + d0)[1] = hi;
  }
}

// ---------------------------------------------------------------------------
extern "C" void kernel_launch(void* const* d_in, const int* in_sizes, int n_in,
                              void* d_out, int out_size, void* d_ws, size_t ws_size,
                              hipStream_t stream) {
  const float* x      = (const float*)d_in[0];   // [4,8192,1024]
  const float* wbatch = (const float*)d_in[1];   // [2048,1024]
  const float* slots  = (const float*)d_in[2];   // [32,1024]
  const float* usage  = (const float*)d_in[3];   // [32]
  float* out = (float*)d_out;

  char* ws = (char*)d_ws;                         // ~132 KB used
  _Float16* snh    = (_Float16*)ws;               // 32*1024 f16 = 64 KB
  _Float16* slotsT = (_Float16*)(ws + 64 * 1024); // 1024*32 f16 = 64 KB
  float*    flagp  = (float*)(ws + 128 * 1024);   // 4 B

  const int T    = in_sizes[1] / 1024;            // 2048 tokens to write
  const int nTok = in_sizes[0] / 1024;            // 32768 query tokens

  msb_update_kernel<<<1, 1024, 0, stream>>>(wbatch, slots, usage, T,
                                            snh, slotsT, flagp);
  msb_retrieve_kernel<<<nTok / 128, 256, 0, stream>>>(x, snh, slotsT, flagp, out);
}
